// CausalSelfAttention_68101001445422
// MI455X (gfx1250) — matrix operand order repaired and checked by reading, not compiled, not run on hardware
//
#include <hip/hip_runtime.h>

#define N_EMBD 1024
#define C3     3072
#define N_HEAD 16
#define HEAD   64
#define SEQ    2048
#define BATCH  2
#define ROWS   (BATCH * SEQ)   // 4096

typedef __attribute__((ext_vector_type(16))) __bf16 v16bf;
typedef __attribute__((ext_vector_type(8)))  float  v8f;

union FragAB { v16bf v; unsigned u[8]; };
union FragC  { v8f   v; float    f[8]; };

#define WMMA_BF16(a, b, c) \
  __builtin_amdgcn_wmma_f32_16x16x32_bf16(false, (a), false, (b), (short)0, (c), false, false)

__device__ __forceinline__ unsigned short f2bf(float x) {
  union { float f; unsigned u; } c; c.f = x;
  unsigned r = c.u + 0x7FFFu + ((c.u >> 16) & 1u);
  return (unsigned short)(r >> 16);
}

// A-matrix 16x32 bf16 fragment (ISA 7.12.2): lane L holds row M=L%16,
// VGPR v holds K pair k = 16*(v/4) + 8*(L/16) + 2*(v%4)  -> two contiguous 16B chunks/lane
__device__ __forceinline__ v16bf load_a16x32(const unsigned short* src, int ld, int lane) {
  int r = lane & 15, h = lane >> 4;
  FragAB f;
  const unsigned short* row = src + (size_t)r * ld;
#pragma unroll
  for (int i = 0; i < 8; ++i) {
    int k = ((i >> 2) << 4) + (h << 3) + ((i & 3) << 1);
    f.u[i] = *(const unsigned*)(row + k);
  }
  return f.v;
}

// B-matrix 32x16 bf16 fragment from [N][K]-transposed source:
// lane L holds col N=L%16, VGPR v holds K pair k = 16*(v/4) + 8*(L/16) + 2*(v%4) -> two contiguous 16B chunks/lane
__device__ __forceinline__ v16bf load_b32x16T(const unsigned short* srcT, int ld, int lane) {
  int n = lane & 15, h = lane >> 4;
  FragAB f;
  const unsigned short* row = srcT + (size_t)n * ld;
#pragma unroll
  for (int i = 0; i < 8; ++i) {
    int k = ((i >> 2) << 4) + (h << 3) + ((i & 3) << 1);
    f.u[i] = *(const unsigned*)(row + k);
  }
  return f.v;
}

// ---------------- conversion kernels ----------------
__global__ void __launch_bounds__(256) cvt_kernel(const float* __restrict__ in,
                                                  unsigned short* __restrict__ out, int n) {
  int i = blockIdx.x * blockDim.x + threadIdx.x;
  int stride = gridDim.x * blockDim.x;
  for (; i < n; i += stride) out[i] = f2bf(in[i]);
}

// outT[n*1024 + k] = bf16(in[k*N + n])  (transpose [1024][N] -> [N][1024]); K fixed = 1024
__global__ void __launch_bounds__(256) cvtT_kernel(const float* __restrict__ in,
                                                   unsigned short* __restrict__ outT,
                                                   int N) {
  int i = blockIdx.x * blockDim.x + threadIdx.x;
  int stride = gridDim.x * blockDim.x;
  int total = N_EMBD * N;
  for (; i < total; i += stride) {
    int n = i >> 10, k = i & 1023;
    outT[i] = f2bf(in[(size_t)k * N + n]);
  }
}

// ---------------- QKV GEMM: [4096,1024] x [1024,3072] + bias ----------------
// 32x64 tile per wave: 2 A-fragments x 4 B-fragments -> 8 accumulators.
// Branch-free inner loop: next-A preload + prefetch are unconditional (reads stay
// inside the mapped workspace; prefetch is speculative per ISA 10.5).
__global__ void __launch_bounds__(128) qkv_kernel(
    const unsigned short* __restrict__ xb,
    const unsigned short* __restrict__ waT,
    const float* __restrict__ b_attn,
    unsigned short* __restrict__ q,
    unsigned short* __restrict__ k,
    unsigned short* __restrict__ vt) {
  int lane = threadIdx.x & 31;
  int wave = blockIdx.x * 4 + (threadIdx.x >> 5);  // 6144 waves
  int mt  = wave / 48;            // 0..127 row tiles of 32
  int n64 = (wave % 48) * 64;     // col group of 64
  const unsigned short* arow0 = xb + (size_t)(mt * 32) * N_EMBD;
  const unsigned short* arow1 = arow0 + (size_t)16 * N_EMBD;
  const unsigned short* bbase = waT + (size_t)n64 * N_EMBD;
  v8f zero = {};
  FragC acc[2][4];
#pragma unroll
  for (int s = 0; s < 2; ++s)
#pragma unroll
    for (int j = 0; j < 4; ++j) acc[s][j].v = zero;

  v16bf a0 = load_a16x32(arow0, N_EMBD, lane);
  v16bf a1 = load_a16x32(arow1, N_EMBD, lane);
  for (int kc = 0; kc < N_EMBD; kc += 32) {
    // issue all 4 B fragments, then preload next A slice (harmless overread on last iter)
    v16bf b0 = load_b32x16T(bbase + (size_t)0 * N_EMBD + kc, N_EMBD, lane);
    v16bf b1 = load_b32x16T(bbase + (size_t)16 * N_EMBD + kc, N_EMBD, lane);
    v16bf b2 = load_b32x16T(bbase + (size_t)32 * N_EMBD + kc, N_EMBD, lane);
    v16bf b3 = load_b32x16T(bbase + (size_t)48 * N_EMBD + kc, N_EMBD, lane);
    v16bf na0 = load_a16x32(arow0 + kc + 32, N_EMBD, lane);
    v16bf na1 = load_a16x32(arow1 + kc + 32, N_EMBD, lane);
    __builtin_prefetch(bbase + (size_t)0 * N_EMBD + kc + 32, 0, 3);
    __builtin_prefetch(bbase + (size_t)32 * N_EMBD + kc + 32, 0, 3);
    acc[0][0].v = WMMA_BF16(a0, b0, acc[0][0].v);
    acc[1][0].v = WMMA_BF16(a1, b0, acc[1][0].v);
    acc[0][1].v = WMMA_BF16(a0, b1, acc[0][1].v);
    acc[1][1].v = WMMA_BF16(a1, b1, acc[1][1].v);
    acc[0][2].v = WMMA_BF16(a0, b2, acc[0][2].v);
    acc[1][2].v = WMMA_BF16(a1, b2, acc[1][2].v);
    acc[0][3].v = WMMA_BF16(a0, b3, acc[0][3].v);
    acc[1][3].v = WMMA_BF16(a1, b3, acc[1][3].v);
    a0 = na0; a1 = na1;
  }

  int rl = lane & 15, hf = lane >> 4;
#pragma unroll
  for (int s = 0; s < 2; ++s) {
#pragma unroll
    for (int j = 0; j < 4; ++j) {
      int col = n64 + j * 16 + rl;
      float bias = b_attn[col];
      int seg = col >> 10;       // 0=q,1=k,2=v
      int c   = col & 1023;
      int h = c >> 6, d = c & 63;
#pragma unroll
      for (int r = 0; r < 8; ++r) {
        int row = mt * 32 + s * 16 + r + (hf << 3);
        int b = row >> 11, t = row & 2047;
        unsigned short val = f2bf(acc[s][j].f[r] + bias);
        if (seg == 0)
          q[((size_t)(b * N_HEAD + h) * SEQ + t) * HEAD + d] = val;
        else if (seg == 1)
          k[((size_t)(b * N_HEAD + h) * SEQ + t) * HEAD + d] = val;
        else
          vt[((size_t)(b * N_HEAD + h) * HEAD + d) * SEQ + t] = val;  // V transposed [D][T]
      }
    }
  }
}

// ---------------- flash attention: per-wave 16 queries ----------------
__global__ void __launch_bounds__(128) attn_kernel(
    const unsigned short* __restrict__ q,
    const unsigned short* __restrict__ k,
    const unsigned short* __restrict__ vt,
    unsigned short* __restrict__ y) {
  __shared__ __attribute__((aligned(16))) unsigned short ldsP[4][16][36]; // per-wave P tile, padded
  int lane = threadIdx.x & 31;
  int wid  = threadIdx.x >> 5;
  int qt = blockIdx.x & 31;   // 32 q-supertiles of 64
  int bh = blockIdx.x >> 5;   // 0..31 (b*16+h)
  int b = bh >> 4, h = bh & 15;
  int qbase = qt * 64 + wid * 16;
  const unsigned short* Q = q  + (size_t)bh * SEQ * HEAD;
  const unsigned short* K = k  + (size_t)bh * SEQ * HEAD;
  const unsigned short* V = vt + (size_t)bh * HEAD * SEQ;   // [D][T]
  v16bf aq0 = load_a16x32(Q + (size_t)qbase * HEAD,      HEAD, lane);
  v16bf aq1 = load_a16x32(Q + (size_t)qbase * HEAD + 32, HEAD, lane);
  int rl = lane & 15, hf = lane >> 4;
  v8f zero = {};
  float m[8], l[8];
  FragC o[4];
#pragma unroll
  for (int j = 0; j < 4; ++j) o[j].v = zero;
#pragma unroll
  for (int r = 0; r < 8; ++r) { m[r] = -1e30f; l[r] = 0.f; }
  unsigned short* pbuf = &ldsP[wid][0][0];

  for (int kb = 0; kb < qbase + 16; kb += 32) {
    __builtin_prefetch(K + (size_t)(kb + 32) * HEAD, 0, 3);   // speculative next K-tile
    // S = Q Kt (two 16-col tiles, two 32-dim K chunks each)
    v8f s0 = zero, s1 = zero;
    s0 = WMMA_BF16(aq0, load_b32x16T(K + (size_t)kb * HEAD,             HEAD, lane), s0);
    s0 = WMMA_BF16(aq1, load_b32x16T(K + (size_t)kb * HEAD + 32,        HEAD, lane), s0);
    s1 = WMMA_BF16(aq0, load_b32x16T(K + (size_t)(kb + 16) * HEAD,      HEAD, lane), s1);
    s1 = WMMA_BF16(aq1, load_b32x16T(K + (size_t)(kb + 16) * HEAD + 32, HEAD, lane), s1);
    FragC u0, u1; u0.v = s0; u1.v = s1;
    int n0 = kb + rl, n1 = kb + 16 + rl;
#pragma unroll
    for (int r = 0; r < 8; ++r) {
      int qrow = qbase + r + (hf << 3);
      float x0 = (n0 <= qrow) ? u0.f[r] * 0.125f : -1e30f;  // 1/sqrt(64), causal mask
      float x1 = (n1 <= qrow) ? u1.f[r] * 0.125f : -1e30f;
      float t = fmaxf(x0, x1);
      t = fmaxf(t, __shfl_xor(t, 1, 16));
      t = fmaxf(t, __shfl_xor(t, 2, 16));
      t = fmaxf(t, __shfl_xor(t, 4, 16));
      t = fmaxf(t, __shfl_xor(t, 8, 16));
      float nm   = fmaxf(m[r], t);
      float corr = __expf(m[r] - nm);
      m[r] = nm;
      float p0 = __expf(x0 - nm);
      float p1 = __expf(x1 - nm);
      float s = p0 + p1;
      s += __shfl_xor(s, 1, 16);
      s += __shfl_xor(s, 2, 16);
      s += __shfl_xor(s, 4, 16);
      s += __shfl_xor(s, 8, 16);
      l[r] = l[r] * corr + s;
#pragma unroll
      for (int j = 0; j < 4; ++j) o[j].f[r] *= corr;
      int prow = r + (hf << 3);
      pbuf[prow * 36 + rl]      = f2bf(p0);   // C-layout -> LDS (row-major P tile)
      pbuf[prow * 36 + 16 + rl] = f2bf(p1);
    }
    asm volatile("s_wait_dscnt 0x0" ::: "memory");   // intra-wave LDS transpose fence
    v16bf pf = load_a16x32(pbuf, 36, lane);          // re-read as A-fragment
#pragma unroll
    for (int j = 0; j < 4; ++j)
      o[j].v = WMMA_BF16(pf,
          load_b32x16T(V + (size_t)(j * 16) * SEQ + kb, SEQ, lane),  // [dim][key], pairs contiguous
          o[j].v);
  }

  // epilogue: normalize and write y[b*T+q][h*64+dim] as bf16
#pragma unroll
  for (int j = 0; j < 4; ++j) {
    int col = h * HEAD + j * 16 + rl;
#pragma unroll
    for (int r = 0; r < 8; ++r) {
      int qrow = qbase + r + (hf << 3);
      y[(size_t)(b * SEQ + qrow) * N_EMBD + col] = f2bf(o[j].f[r] / l[r]);
    }
  }
}

// ---------------- output projection: [4096,1024] x [1024,1024] + bias -> f32 ----------------
// 32x64 tile per wave, same branch-free pipelining as qkv.
__global__ void __launch_bounds__(128) proj_kernel(
    const unsigned short* __restrict__ yb,
    const unsigned short* __restrict__ wpT,
    const float* __restrict__ b_proj,
    float* __restrict__ out) {
  int lane = threadIdx.x & 31;
  int wave = blockIdx.x * 4 + (threadIdx.x >> 5);  // 2048 waves
  int mt  = wave >> 4;           // 0..127 row tiles of 32
  int n64 = (wave & 15) * 64;
  const unsigned short* arow0 = yb + (size_t)(mt * 32) * N_EMBD;
  const unsigned short* arow1 = arow0 + (size_t)16 * N_EMBD;
  const unsigned short* bbase = wpT + (size_t)n64 * N_EMBD;
  v8f zero = {};
  FragC acc[2][4];
#pragma unroll
  for (int s = 0; s < 2; ++s)
#pragma unroll
    for (int j = 0; j < 4; ++j) acc[s][j].v = zero;

  v16bf a0 = load_a16x32(arow0, N_EMBD, lane);
  v16bf a1 = load_a16x32(arow1, N_EMBD, lane);
  for (int kc = 0; kc < N_EMBD; kc += 32) {
    v16bf b0 = load_b32x16T(bbase + (size_t)0 * N_EMBD + kc, N_EMBD, lane);
    v16bf b1 = load_b32x16T(bbase + (size_t)16 * N_EMBD + kc, N_EMBD, lane);
    v16bf b2 = load_b32x16T(bbase + (size_t)32 * N_EMBD + kc, N_EMBD, lane);
    v16bf b3 = load_b32x16T(bbase + (size_t)48 * N_EMBD + kc, N_EMBD, lane);
    v16bf na0 = load_a16x32(arow0 + kc + 32, N_EMBD, lane);
    v16bf na1 = load_a16x32(arow1 + kc + 32, N_EMBD, lane);
    __builtin_prefetch(bbase + (size_t)0 * N_EMBD + kc + 32, 0, 3);
    __builtin_prefetch(bbase + (size_t)32 * N_EMBD + kc + 32, 0, 3);
    acc[0][0].v = WMMA_BF16(a0, b0, acc[0][0].v);
    acc[1][0].v = WMMA_BF16(a1, b0, acc[1][0].v);
    acc[0][1].v = WMMA_BF16(a0, b1, acc[0][1].v);
    acc[1][1].v = WMMA_BF16(a1, b1, acc[1][1].v);
    acc[0][2].v = WMMA_BF16(a0, b2, acc[0][2].v);
    acc[1][2].v = WMMA_BF16(a1, b2, acc[1][2].v);
    acc[0][3].v = WMMA_BF16(a0, b3, acc[0][3].v);
    acc[1][3].v = WMMA_BF16(a1, b3, acc[1][3].v);
    a0 = na0; a1 = na1;
  }

  int rl = lane & 15, hf = lane >> 4;
#pragma unroll
  for (int s = 0; s < 2; ++s) {
#pragma unroll
    for (int j = 0; j < 4; ++j) {
      int col = n64 + j * 16 + rl;
      float bias = b_proj[col];
#pragma unroll
      for (int r = 0; r < 8; ++r) {
        int row = mt * 32 + s * 16 + r + (hf << 3);
        out[(size_t)row * N_EMBD + col] = acc[s][j].f[r] + bias;
      }
    }
  }
}

extern "C" void kernel_launch(void* const* d_in, const int* in_sizes, int n_in,
                              void* d_out, int out_size, void* d_ws, size_t ws_size,
                              hipStream_t stream) {
  const float* x      = (const float*)d_in[0];
  const float* w_attn = (const float*)d_in[1];
  const float* b_attn = (const float*)d_in[2];
  const float* w_proj = (const float*)d_in[3];
  const float* b_proj = (const float*)d_in[4];

  char* ws = (char*)d_ws;
  size_t off = 0;
  unsigned short* xb  = (unsigned short*)(ws + off); off += (size_t)ROWS * N_EMBD * 2;               // 8 MB
  unsigned short* waT = (unsigned short*)(ws + off); off += (size_t)C3 * N_EMBD * 2;                 // 6 MB
  unsigned short* wpT = (unsigned short*)(ws + off); off += (size_t)N_EMBD * N_EMBD * 2;             // 2 MB
  unsigned short* qb  = (unsigned short*)(ws + off); off += (size_t)BATCH * N_HEAD * SEQ * HEAD * 2; // 8 MB
  unsigned short* kb  = (unsigned short*)(ws + off); off += (size_t)BATCH * N_HEAD * SEQ * HEAD * 2; // 8 MB
  unsigned short* vtb = (unsigned short*)(ws + off); off += (size_t)BATCH * N_HEAD * HEAD * SEQ * 2; // 8 MB
  unsigned short* yb  = (unsigned short*)(ws + off); off += (size_t)ROWS * N_EMBD * 2;               // 8 MB

  cvt_kernel <<<2048, 256, 0, stream>>>(x, xb, ROWS * N_EMBD);
  cvtT_kernel<<<2048, 256, 0, stream>>>(w_attn, waT, C3);
  cvtT_kernel<<<1024, 256, 0, stream>>>(w_proj, wpT, N_EMBD);

  qkv_kernel <<<1536, 128, 0, stream>>>(xb, waT, b_attn, qb, kb, vtb);
  attn_kernel<<<1024, 128, 0, stream>>>(qb, kb, vtb, yb);
  proj_kernel<<< 512, 128, 0, stream>>>(yb, wpT, b_proj, (float*)d_out);
}
